// final_conv_select_net_3264175144993
// MI455X (gfx1250) — compile-verified, matrix operand order checked by reading, not run on hardware
//
#include <hip/hip_runtime.h>
#include <math.h>

// MI455X / gfx1250: wave32, WMMA 16x16x32 f16 -> f32 accumulate.
typedef _Float16 f16;
typedef __attribute__((ext_vector_type(16))) _Float16 v16h;
typedef __attribute__((ext_vector_type(8)))  float    v8f;

#define NEXP 64        // experts (K in reference)
#define DIN  512
#define D1   256
#define D2   128
#define D3   64
#define BM   64        // rows of x per workgroup
#define NTHREADS 256   // 8 waves of 32

// Padded LDS row strides (halfs): +8 halfs (16B) so consecutive rows land
// 4 banks apart -> 16 lanes x b128 (4 banks each) tile all 64 banks once.
#define XS_LD (DIN + 8)   // 520
#define H1_LD (D1 + 8)    // 264
#define H2_LD (D2 + 8)    // 136
#define H3_LD (D3 + 8)    // 72

#define XS_BYTES ((size_t)BM * XS_LD * 2)   // 66560
#define H1_BYTES ((size_t)BM * H1_LD * 2)   // 33792
#define H2_BYTES ((size_t)BM * H2_LD * 2)   // 17408
#define SMEM_BYTES (XS_BYTES + H1_BYTES)    // ~98KB; H2/H3 alias dead Xs

// Swizzled f16 weight arena layout in d_ws (halfs).
#define W1S_HALFS ((size_t)NEXP * DIN * D1)
#define W2S_HALFS ((size_t)NEXP * D1 * D2)
#define W3S_HALFS ((size_t)NEXP * D2 * D3)
#define WS_NEEDED_BYTES ((W1S_HALFS + W2S_HALFS + W3S_HALFS) * 2)

__device__ __forceinline__ v8f wmma_f16(v16h a, v16h b, v8f c) {
  // (neg_a, A, neg_b, B, c_mod, C, reuse_a, reuse_b)
  return __builtin_amdgcn_wmma_f32_16x16x32_f16(false, a, false, b, (short)0, c,
                                                false, false);
}

// A fragment (16x32) from row-major f16 LDS, padded row stride `ld` (halfs).
// Lane L: row M=L&15, kb = k0 + 8*(L>>4); halfs h -> K = kb+16*(h>>3)+(h&7).
// Two contiguous 16B runs per lane -> 2x ds_load_b128, conflict-free.
__device__ __forceinline__ v16h load_a_lds(const f16* base, int ld, int mt,
                                           int k0, int lane) {
  const f16* p = base + (mt * 16 + (lane & 15)) * ld + k0 + ((lane >> 4) << 3);
  v16h a;
#pragma unroll
  for (int h = 0; h < 8; ++h) { a[h] = p[h]; a[h + 8] = p[h + 16]; }
  return a;
}

// Fallback B fragment (32x16) gathered from row-major f32 weights, ldb = d_out.
__device__ __forceinline__ v16h load_b_g32(const float* __restrict__ W, int ldb,
                                           int k0, int n0, int lane) {
  const float* p = W + (size_t)(k0 + ((lane >> 4) << 3)) * ldb + (n0 + (lane & 15));
  v16h b;
#pragma unroll
  for (int h = 0; h < 8; ++h) {
    b[h]     = (f16)p[(size_t)h * ldb];
    b[h + 8] = (f16)p[(size_t)(h + 16) * ldb];
  }
  return b;
}

// Fast path: pre-swizzled f16 B fragments; lane L's 16 halfs contiguous (32B),
// 1KB block per fragment, fully coalesced across the wave.
__device__ __forceinline__ v16h load_b_swz(const f16* __restrict__ Wswz,
                                           int kc, int nt, int nt_cnt, int lane) {
  const f16* p = Wswz + (((size_t)(kc * nt_cnt + nt) * 32) + lane) * 16;
  return *(const v16h*)p;
}

// C/D layout: VGPR r holds M = mt*16 + 8*(lane>>4) + r, N = nt*16 + (lane&15).
__device__ __forceinline__ void store_bias_relu(f16* H, int ld, int mt, int nt,
                                                v8f acc, float bias, int lane) {
  const int col = nt * 16 + (lane & 15);
  const int r0  = mt * 16 + ((lane >> 4) << 3);
#pragma unroll
  for (int r = 0; r < 8; ++r) {
    float v = acc[r] + bias;
    v = v > 0.0f ? v : 0.0f;
    H[(r0 + r) * ld + col] = (f16)v;
  }
}

// ---------------------------------------------------------------------------
// Prep: convert W[e][din][dout] f32 -> f16 in WMMA B-fragment order.
// Thread t = ((e*kc_cnt + kc)*nt_cnt + nt)*32 + lane; writes dst[t*16 .. +15].
// Grids sized exactly (all counts are multiples of 256) -> no bounds check.
// ---------------------------------------------------------------------------
__global__ __launch_bounds__(256)
void swizzle_weights(const float* __restrict__ W, f16* __restrict__ dst,
                     int din, int dout) {
  const int t    = blockIdx.x * 256 + threadIdx.x;
  const int lane = t & 31;
  const int idx  = t >> 5;
  const int nt_cnt = dout >> 4;
  const int kc_cnt = din >> 5;
  const int nt   = idx % nt_cnt;
  const int rest = idx / nt_cnt;
  const int kc   = rest % kc_cnt;
  const int e    = rest / kc_cnt;

  const float* src = W + (size_t)e * din * dout;
  const int kb  = kc * 32 + ((lane >> 4) << 3);
  const int col = nt * 16 + (lane & 15);
  v16h v;
#pragma unroll
  for (int h = 0; h < 8; ++h) {
    v[h]     = (f16)src[(size_t)(kb + h) * dout + col];
    v[h + 8] = (f16)src[(size_t)(kb + 16 + h) * dout + col];
  }
  *(v16h*)(dst + (size_t)t * 16) = v;
}

// ---------------------------------------------------------------------------
// Main fused 4-layer expert MLP.
// ---------------------------------------------------------------------------
template <bool SWZ>
__global__ __launch_bounds__(NTHREADS)
void expert_mlp_kernel(const float* __restrict__ x,
                       const float* __restrict__ W1, const float* __restrict__ b1,
                       const float* __restrict__ W2, const float* __restrict__ b2,
                       const float* __restrict__ W3, const float* __restrict__ b3,
                       const float* __restrict__ W4, const float* __restrict__ b4,
                       const f16* __restrict__ W1s, const f16* __restrict__ W2s,
                       const f16* __restrict__ W3s,
                       float* __restrict__ out) {
  __shared__ __align__(16) unsigned char smem[SMEM_BYTES];
  f16* Xs = (f16*)smem;                 // [BM][XS_LD]  layer-1 input (f16)
  f16* H1 = (f16*)(smem + XS_BYTES);    // [BM][H1_LD]
  f16* H2 = (f16*)smem;                 // [BM][H2_LD]  aliases dead Xs
  f16* H3 = (f16*)(smem + H2_BYTES);    // [BM][H3_LD]  aliases dead Xs

  const int tid  = threadIdx.x;
  const int lane = tid & 31;
  const int wave = tid >> 5;
  const int e    = blockIdx.x;            // expert
  const int m0   = blockIdx.y * BM;       // row-tile base

  const float* xe  = x  + (size_t)m0 * (NEXP * DIN) + (size_t)e * DIN;
  const float* W1e = W1 + (size_t)e * DIN * D1;
  const float* W2e = W2 + (size_t)e * D1 * D2;
  const float* W3e = W3 + (size_t)e * D2 * D3;
  const float* W4e = W4 + (size_t)e * D3;
  const float* b1e = b1 + (size_t)e * D1;
  const float* b2e = b2 + (size_t)e * D2;
  const float* b3e = b3 + (size_t)e * D3;
  const f16* W1se = SWZ ? W1s + (size_t)e * DIN * D1 : nullptr;
  const f16* W2se = SWZ ? W2s + (size_t)e * D1 * D2  : nullptr;
  const f16* W3se = SWZ ? W3s + (size_t)e * D2 * D3  : nullptr;

  // ---- Stage x tile -> LDS as f16 (convert ONCE per workgroup) -----------
  // Coalesced: 128 consecutive threads cover one 512-float row via b128s.
#pragma unroll 4
  for (int idx = tid * 4; idx < BM * DIN; idx += NTHREADS * 4) {
    const int row = idx >> 9;          // / DIN
    const int col = idx & (DIN - 1);
    const float4 v = *(const float4*)(xe + (size_t)row * (NEXP * DIN) + col);
    f16* d = &Xs[row * XS_LD + col];
    d[0] = (f16)v.x; d[1] = (f16)v.y; d[2] = (f16)v.z; d[3] = (f16)v.w;
  }
  __syncthreads();

  // ---- Layer 1: [64,512] x [512,256] -> H1, relu -------------------------
  // Each wave: N-tiles {2w, 2w+1}, all 4 M-tiles (8 f32 accumulators).
  {
    v8f acc[4][2] = {};
    const int nt0 = wave * 2;
#pragma unroll 1
    for (int k0 = 0; k0 < DIN; k0 += 32) {
      v16h bf0, bf1;
      if (SWZ) {
        bf0 = load_b_swz(W1se, k0 >> 5, nt0 + 0, D1 >> 4, lane);
        bf1 = load_b_swz(W1se, k0 >> 5, nt0 + 1, D1 >> 4, lane);
      } else {
        bf0 = load_b_g32(W1e, D1, k0, (nt0 + 0) * 16, lane);
        bf1 = load_b_g32(W1e, D1, k0, (nt0 + 1) * 16, lane);
      }
#pragma unroll
      for (int mt = 0; mt < 4; ++mt) {
        v16h af = load_a_lds(Xs, XS_LD, mt, k0, lane);
        acc[mt][0] = wmma_f16(af, bf0, acc[mt][0]);
        acc[mt][1] = wmma_f16(af, bf1, acc[mt][1]);
      }
    }
    __syncthreads();   // all Xs reads done before H1 stores race with nothing,
                       // and before H2/H3 (aliasing Xs) are written in layer 2.
#pragma unroll
    for (int j = 0; j < 2; ++j) {
      const float bias = b1e[(nt0 + j) * 16 + (lane & 15)];
#pragma unroll
      for (int mt = 0; mt < 4; ++mt)
        store_bias_relu(H1, H1_LD, mt, nt0 + j, acc[mt][j], bias, lane);
    }
  }
  __syncthreads();

  // ---- Layer 2: [64,256] x [256,128] -> H2, relu -------------------------
  {
    v8f acc[4] = {};
    const int nt = wave;
#pragma unroll 1
    for (int k0 = 0; k0 < D1; k0 += 32) {
      v16h bf = SWZ ? load_b_swz(W2se, k0 >> 5, nt, D2 >> 4, lane)
                    : load_b_g32(W2e, D2, k0, nt * 16, lane);
#pragma unroll
      for (int mt = 0; mt < 4; ++mt) {
        v16h af = load_a_lds(H1, H1_LD, mt, k0, lane);
        acc[mt] = wmma_f16(af, bf, acc[mt]);
      }
    }
    const float bias = b2e[nt * 16 + (lane & 15)];
#pragma unroll
    for (int mt = 0; mt < 4; ++mt)
      store_bias_relu(H2, H2_LD, mt, nt, acc[mt], bias, lane);
  }
  __syncthreads();

  // ---- Layer 3: [64,128] x [128,64] -> H3, relu --------------------------
  {
    v8f acc[2] = {};
    const int nt = wave & 3;
    const int mb = (wave >> 2) * 2;
#pragma unroll 1
    for (int k0 = 0; k0 < D2; k0 += 32) {
      v16h bf = SWZ ? load_b_swz(W3se, k0 >> 5, nt, D3 >> 4, lane)
                    : load_b_g32(W3e, D3, k0, nt * 16, lane);
#pragma unroll
      for (int m = 0; m < 2; ++m) {
        v16h af = load_a_lds(H2, H2_LD, mb + m, k0, lane);
        acc[m] = wmma_f16(af, bf, acc[m]);
      }
    }
    const float bias = b3e[nt * 16 + (lane & 15)];
#pragma unroll
    for (int m = 0; m < 2; ++m)
      store_bias_relu(H3, H3_LD, mb + m, nt, acc[m], bias, lane);
  }
  __syncthreads();

  // ---- Layer 4: [64,64] x [64,1] + sigmoid (scalar, tiny) ----------------
  if (tid < BM) {
    const f16* h = &H3[tid * H3_LD];
    float acc = b4[e];
#pragma unroll
    for (int j = 0; j < D3; ++j) acc += (float)h[j] * W4e[j];
    out[(size_t)(m0 + tid) * NEXP + e] = 1.0f / (1.0f + __expf(-acc));
  }
}

extern "C" void kernel_launch(void* const* d_in, const int* in_sizes, int n_in,
                              void* d_out, int out_size, void* d_ws, size_t ws_size,
                              hipStream_t stream) {
  (void)in_sizes; (void)n_in; (void)out_size;
  const float* x  = (const float*)d_in[0];
  const float* W1 = (const float*)d_in[1];
  const float* b1 = (const float*)d_in[2];
  const float* W2 = (const float*)d_in[3];
  const float* b2 = (const float*)d_in[4];
  const float* W3 = (const float*)d_in[5];
  const float* b3 = (const float*)d_in[6];
  const float* W4 = (const float*)d_in[7];
  const float* b4 = (const float*)d_in[8];
  float* out = (float*)d_out;

  const bool use_swz = (d_ws != nullptr) && (ws_size >= WS_NEEDED_BYTES);
  dim3 grid(NEXP, 8192 / BM);   // 64 experts x 128 row-tiles

  if (use_swz) {
    f16* W1s = (f16*)d_ws;
    f16* W2s = W1s + W1S_HALFS;
    f16* W3s = W2s + W2S_HALFS;
    // Exact-size grids: counts are multiples of 256.
    swizzle_weights<<<(int)(W1S_HALFS / 16 / 256), 256, 0, stream>>>(W1, W1s, DIN, D1);
    swizzle_weights<<<(int)(W2S_HALFS / 16 / 256), 256, 0, stream>>>(W2, W2s, D1, D2);
    swizzle_weights<<<(int)(W3S_HALFS / 16 / 256), 256, 0, stream>>>(W3, W3s, D2, D3);
    expert_mlp_kernel<true><<<grid, NTHREADS, 0, stream>>>(
        x, W1, b1, W2, b2, W3, b3, W4, b4, W1s, W2s, W3s, out);
  } else {
    expert_mlp_kernel<false><<<grid, NTHREADS, 0, stream>>>(
        x, W1, b1, W2, b2, W3, b3, W4, b4, nullptr, nullptr, nullptr, out);
  }
}
